// APPNPNet_9208409883100
// MI455X (gfx1250) — compile-verified, matrix operand order checked by reading, not run on hardware
//
#include <hip/hip_runtime.h>

// ---- CDNA5 WMMA types ------------------------------------------------------
typedef __attribute__((ext_vector_type(16))) _Float16 v16h;
typedef __attribute__((ext_vector_type(8)))  float    v8f;

#define ALPHA 0.1f
#define KPROP 10
#define INC   512
#define HIDC  256
#define OUTC  64

// K offset inside a 16-bit A/B fragment (CDNA5 ISA 7.12.2):
//   VGPR v in 0..3 -> K = 2v,2v+1 (lanes 0-15), +8 for lanes 16-31
//   VGPR v in 4..7 -> K = 16+2(v-4),..  (+8 for upper half-wave)
// element e (0..15) of v16h: v = e/2, odd = e&1.
__device__ __forceinline__ int kofs(int e, int half) {
  int v = e >> 1;
  int base = (v < 4) ? (v * 2) : (16 + (v - 4) * 2);
  return base + (e & 1) + (half << 3);
}

// ---- weight packing: f32 row-major [K][N] -> f16 fragment order ------------
// layout: [ntile][ktile][lane][e], 16 f16 (32B) contiguous per lane per ktile.
__global__ void pack_w1(const float* __restrict__ w1, _Float16* __restrict__ w1p) {
  int i = blockIdx.x * blockDim.x + threadIdx.x;      // 16*16*32*16 = 131072
  int e = i & 15, lane = (i >> 4) & 31, kb = (i >> 9) & 15, nt = i >> 13;
  int k = kb * 32 + kofs(e, lane >> 4);
  int n = nt * 16 + (lane & 15);
  w1p[i] = (_Float16)w1[k * HIDC + n];
}

__global__ void pack_w2(const float* __restrict__ w2, _Float16* __restrict__ w2p) {
  int i = blockIdx.x * blockDim.x + threadIdx.x;      // 4*8*32*16 = 16384
  int e = i & 15, lane = (i >> 4) & 31, kb = (i >> 9) & 7, nt = i >> 12;
  int k = kb * 32 + kofs(e, lane >> 4);
  int n = nt * 16 + (lane & 15);
  w2p[i] = (_Float16)w2[k * OUTC + n];
}

// ---- degree / normalization ------------------------------------------------
__global__ void deg_init(float* __restrict__ deg, int n) {
  int i = blockIdx.x * blockDim.x + threadIdx.x;
  if (i < n) deg[i] = 1.0f;                    // self-loop
}
__global__ void deg_accum(const int* __restrict__ dst, float* __restrict__ deg, int nedge) {
  int e = blockIdx.x * blockDim.x + threadIdx.x;
  if (e < nedge) atomicAdd(deg + dst[e], 1.0f);
}
__global__ void deg_rsqrt(float* __restrict__ deg, int n) {
  int i = blockIdx.x * blockDim.x + threadIdx.x;
  if (i < n) { float d = deg[i]; deg[i] = d > 0.f ? rsqrtf(d) : 0.f; }
}
__global__ void edge_norm(const int* __restrict__ src, const int* __restrict__ dst,
                          const float* __restrict__ dinv, float* __restrict__ normE, int nedge) {
  int e = blockIdx.x * blockDim.x + threadIdx.x;
  if (e < nedge) normE[e] = (1.0f - ALPHA) * dinv[src[e]] * dinv[dst[e]];
}

// ---- fused MLP: h0 = relu(x@W1+b1)@W2+b2, WMMA f16->f32 --------------------
// 2 waves / block, 16 rows / wave. GEMM1 result transposed through LDS.
__global__ __launch_bounds__(64) void mlp_fused(
    const float* __restrict__ x, const float* __restrict__ b1,
    const float* __restrict__ b2, const _Float16* __restrict__ w1p,
    const _Float16* __restrict__ w2p, float* __restrict__ h0, int n) {
  __shared__ float hid[2][16 * 257];                  // stride 257: conflict-free
  const int wave = threadIdx.x >> 5;
  const int lane = threadIdx.x & 31;
  const int hf   = lane >> 4;
  const int lm   = lane & 15;
  const int rowbase = blockIdx.x * 32 + wave * 16;

  int r = rowbase + lm; if (r >= n) r = n - 1;        // clamp: keep EXEC all-1s
  const float* xr = x + (size_t)r * INC;

  // A fragments for the whole K=512 (16 frags, lane = row M)
  v16h a[16];
#pragma unroll
  for (int kb = 0; kb < 16; ++kb) {
#pragma unroll
    for (int e = 0; e < 16; ++e)
      a[kb][e] = (_Float16)xr[kb * 32 + kofs(e, hf)];
  }

  float* hrow = &hid[wave][0];
#pragma unroll 1
  for (int nt = 0; nt < 16; ++nt) {                   // 16 N-tiles of HIDC=256
    v8f c = {};
#pragma unroll
    for (int kb = 0; kb < 16; ++kb) {
      v16h b = *(const v16h*)(w1p + (size_t)((nt * 16 + kb) * 32 + lane) * 16);
      c = __builtin_amdgcn_wmma_f32_16x16x32_f16(false, a[kb], false, b,
                                                 (short)0, c, false, false);
    }
    float bias = b1[nt * 16 + lm];
#pragma unroll
    for (int m8 = 0; m8 < 8; ++m8) {                  // C: lane=col, vgpr=row
      float v = c[m8] + bias;
      hrow[(m8 + hf * 8) * 257 + nt * 16 + lm] = v > 0.f ? v : 0.f;
    }
  }
  __syncthreads();

  // A fragments for GEMM2 from LDS (K=256 -> 8 frags), lane = row M again
  v16h a2[8];
#pragma unroll
  for (int kb = 0; kb < 8; ++kb) {
#pragma unroll
    for (int e = 0; e < 16; ++e)
      a2[kb][e] = (_Float16)hrow[lm * 257 + kb * 32 + kofs(e, hf)];
  }

#pragma unroll 1
  for (int nt = 0; nt < 4; ++nt) {                    // 4 N-tiles of OUTC=64
    v8f c = {};
#pragma unroll
    for (int kb = 0; kb < 8; ++kb) {
      v16h b = *(const v16h*)(w2p + (size_t)((nt * 8 + kb) * 32 + lane) * 16);
      c = __builtin_amdgcn_wmma_f32_16x16x32_f16(false, a2[kb], false, b,
                                                 (short)0, c, false, false);
    }
    float bias = b2[nt * 16 + lm];
#pragma unroll
    for (int m8 = 0; m8 < 8; ++m8) {
      int row = rowbase + m8 + hf * 8;
      if (row < n) h0[(size_t)row * OUTC + nt * 16 + lm] = c[m8] + bias;
    }
  }
}

// ---- propagation: out = 0.1*h0 + 0.9*(dinv^2*hc  +  sum_e norm_e*hc[src]) --
__global__ void prop_init(const float* __restrict__ h0, const float* __restrict__ hc,
                          const float* __restrict__ dinv, float* __restrict__ out,
                          int total) {
  int i = blockIdx.x * blockDim.x + threadIdx.x;
  if (i >= total) return;
  float s = dinv[i >> 6];                             // self-loop norm = dinv^2
  out[i] = ALPHA * h0[i] + (1.0f - ALPHA) * s * s * hc[i];
}

__global__ void prop_scatter(const int* __restrict__ src, const int* __restrict__ dst,
                             const float* __restrict__ normE, const float* __restrict__ hc,
                             float* __restrict__ out, int nedge) {
  int t = blockIdx.x * blockDim.x + threadIdx.x;
  int e = t >> 6;                                     // 64 channels per edge
  if (e >= nedge) return;
  int c = t & 63;
  float v = normE[e] * hc[(size_t)src[e] * OUTC + c]; // L2-resident gather
  atomicAdd(out + (size_t)dst[e] * OUTC + c, v);      // L2-resident f32 atomic
}

// ---- host side -------------------------------------------------------------
extern "C" void kernel_launch(void* const* d_in, const int* in_sizes, int n_in,
                              void* d_out, int out_size, void* d_ws, size_t ws_size,
                              hipStream_t stream) {
  const float* x  = (const float*)d_in[0];
  const int*   ei = (const int*)d_in[1];
  const float* w1 = (const float*)d_in[2];
  const float* b1 = (const float*)d_in[3];
  const float* w2 = (const float*)d_in[4];
  const float* b2 = (const float*)d_in[5];
  const int n     = in_sizes[0] / INC;
  const int nedge = in_sizes[1] / 2;
  const int* srcp = ei;
  const int* dstp = ei + nedge;
  float* out = (float*)d_out;

  // workspace carve (h0 | hA | dinv | normE | packed weights), 256B aligned
  char* ws = (char*)d_ws;
  size_t off = 0;
  auto carve = [&](size_t bytes) -> void* {
    void* p = ws + off;
    off = (off + bytes + 255) & ~(size_t)255;
    return p;
  };
  float*    h0    = (float*)carve((size_t)n * OUTC * 4);
  float*    hA    = (float*)carve((size_t)n * OUTC * 4);
  float*    dinv  = (float*)carve((size_t)n * 4);
  float*    normE = (float*)carve((size_t)nedge * 4);
  _Float16* w1p   = (_Float16*)carve((size_t)INC * HIDC * 2);
  _Float16* w2p   = (_Float16*)carve((size_t)HIDC * OUTC * 2);
  (void)ws_size; (void)n_in; (void)out_size;

  pack_w1<<<(INC * HIDC) / 256, 256, 0, stream>>>(w1, w1p);
  pack_w2<<<(HIDC * OUTC) / 256, 256, 0, stream>>>(w2, w2p);

  deg_init <<<(n + 255) / 256, 256, 0, stream>>>(dinv, n);
  deg_accum<<<(nedge + 255) / 256, 256, 0, stream>>>(dstp, dinv, nedge);
  deg_rsqrt<<<(n + 255) / 256, 256, 0, stream>>>(dinv, n);
  edge_norm<<<(nedge + 255) / 256, 256, 0, stream>>>(srcp, dstp, dinv, normE, nedge);

  mlp_fused<<<(n + 31) / 32, 64, 0, stream>>>(x, b1, b2, w1p, w2p, h0, n);

  const int total = n * OUTC;
  const long et = (long)nedge * OUTC;
  for (int k = 0; k < KPROP; ++k) {
    // ping-pong so step 9 (odd) lands in d_out
    const float* cur = (k == 0) ? h0 : ((k & 1) ? hA : out);
    float*       nxt = (k & 1) ? out : hA;
    prop_init   <<<(total + 255) / 256, 256, 0, stream>>>(h0, cur, dinv, nxt, total);
    prop_scatter<<<(unsigned)((et + 255) / 256), 256, 0, stream>>>(srcp, dstp, normE,
                                                                   cur, nxt, nedge);
  }
}
